// LSTMGCNPMAGbp_66434554134670
// MI455X (gfx1250) — matrix operand order repaired and checked by reading, not compiled, not run on hardware
//
#include <hip/hip_runtime.h>
#include <cmath>

// ---------------------------------------------------------------------------
// Fused LSTM + GCN + PNA + predict for MI455X (gfx1250, wave32, WMMA).
// One workgroup (256 thr = 8 waves) per batch group b.
//  - LSTM gates: v_wmma_f32_16x16x32_bf16, W_ih/W_hh B-fragments in VGPRs
//  - x_t gather: async global->LDS copies (ASYNCcnt), double buffered
//  - activations: branch-free v_exp/v_rcp formulations
// ---------------------------------------------------------------------------

typedef __attribute__((ext_vector_type(16))) __bf16 v16bf;
typedef __attribute__((ext_vector_type(8)))  float  v8f;
typedef __attribute__((address_space(3)))    char   lds_char_t;

#define B_GROUPS 1024
#define M_MEM    16
#define K_NBR    32
#define L_SEQ    50
#define E_DIM    64

#define XSF   68      // f32 row stride for x LDS tiles (padded, 16B-aligned rows)
#define XS    72      // bf16 row stride for h LDS tile (padded)
#define GSTR  260     // f32 row stride for gates LDS (padded)

// ---- branch-free activations (v_exp_f32 + v_rcp_f32, saturate correctly) ---
__device__ __forceinline__ float sigmoidf_(float x) {
    return __builtin_amdgcn_rcpf(1.0f + __expf(-x));
}
__device__ __forceinline__ float tanhf_(float x) {
    return 1.0f - 2.0f * __builtin_amdgcn_rcpf(1.0f + __expf(2.0f * x));
}

// ---- async global -> LDS copy (16B per lane), tracked by ASYNCcnt ----------
__device__ __forceinline__ void async_copy_b128(void* lds_dst, const void* gsrc) {
    const unsigned ldsoff = (unsigned)(unsigned long long)(lds_char_t*)lds_dst;
    asm volatile("global_load_async_to_lds_b128 %0, %1, off"
                 :: "v"(ldsoff), "v"(gsrc) : "memory");
}
__device__ __forceinline__ void wait_async_le1() {
    asm volatile("s_wait_asynccnt 0x1" ::: "memory");
}
__device__ __forceinline__ void wait_async_0() {
    asm volatile("s_wait_asynccnt 0x0" ::: "memory");
}

// A-fragment (16x32 bf16) built from an f32 LDS tile [16][XSF].
// ISA layout: lanes 0-15 row=lane, klo=0; lanes 16-31 same rows, klo=8;
// elems 0..7 -> K=klo+0..7, elems 8..15 -> K=16+klo+0..7.
__device__ __forceinline__ v16bf load_a_frag_f32(const float* X, int kbase, int lane) {
    const int row = lane & 15;
    const int klo = (lane & 16) ? 8 : 0;
    const float* p = X + row * XSF + kbase + klo;
    v16bf a;
#pragma unroll
    for (int e = 0; e < 8; ++e) a[e] = (__bf16)p[e];
#pragma unroll
    for (int e = 0; e < 8; ++e) a[8 + e] = (__bf16)p[16 + e];
    return a;
}

// A-fragment from a bf16 LDS tile [16][XS] (used for h).
__device__ __forceinline__ v16bf load_a_frag_bf16(const __bf16* X, int kbase, int lane) {
    const int row = lane & 15;
    const int klo = (lane & 16) ? 8 : 0;
    const __bf16* p = X + row * XS + kbase + klo;
    v16bf a;
#pragma unroll
    for (int e = 0; e < 8; ++e) a[e] = p[e];
#pragma unroll
    for (int e = 0; e < 8; ++e) a[8 + e] = p[16 + e];
    return a;
}

// B-fragment (32x16 bf16) of W^T where W is row-major [N][K]: B[k][n]=W[n][k].
// Lanes 0-15: col n0+lane, K=kbase+0..7,16..23; lanes 16-31: col, K=kbase+8..15,24..31.
__device__ __forceinline__ v16bf load_bT_frag(const float* W, int rowlen,
                                              int n0, int kbase, int lane) {
    const int n  = n0 + (lane & 15);
    const int k0 = kbase + ((lane & 16) ? 8 : 0);
    const float* p = W + (size_t)n * rowlen + k0;
    v16bf b;
#pragma unroll
    for (int e = 0; e < 16; ++e) b[e] = (__bf16)p[(e & 7) + ((e & 8) ? 16 : 0)];
    return b;
}

__device__ __forceinline__ v8f wmma_bf16(v16bf a, v16bf b, v8f c) {
    return __builtin_amdgcn_wmma_f32_16x16x32_bf16(
        false, a, false, b, (short)0, c, false, false);
}

__global__ __launch_bounds__(256)
void fused_lstm_gcn_pna_pred(
    const int*   members,    const int*   neighbors,
    const int*   act_inputs, const int*   act_seqs,
    const float* act_table,  const float* user_table,
    const float* W_ih, const float* W_hh, const float* b_ih, const float* b_hh,
    const float* gcn_W, const float* gcn_b,
    const float* pna_W1, const float* pna_b1,
    const float* pna_W2, const float* pna_b2,
    const float* pna_W3, const float* pna_b3,
    const float* pred_W1, const float* pred_b1,
    const float* pred_W2, const float* pred_b2,
    float* out)
{
    __shared__ float  x_lds[2][M_MEM * XSF];   // double-buffered x_t (f32, async dst)
    __shared__ __bf16 h_lds[M_MEM * XS];       // recurrent h (bf16 A-matrix)
    __shared__ float  gates[M_MEM * GSTR];
    __shared__ float  fus  [M_MEM * E_DIM];
    __shared__ int    seq_lds[M_MEM * L_SEQ];  // staged act_seqs slice
    __shared__ float  feats[4 * E_DIM];
    __shared__ float  v1   [E_DIM];
    __shared__ float  v2   [E_DIM];
    __shared__ float  grp  [E_DIM];
    __shared__ float  acte [E_DIM];
    __shared__ float  hred [8];

    const int tid  = threadIdx.x;
    const int lane = tid & 31;
    const int wave = tid >> 5;
    const int gb   = blockIdx.x;

    // elementwise mapping: thread -> (member m, 4 consecutive cols at e4)
    const int m  = tid >> 4;
    const int e4 = (tid & 15) * 4;

    // ---- preload loop-invariant weight B-fragments into VGPRs -------------
    // Wave w owns gate-column tiles {w, w+8}; K split into 2 fragments.
    v16bf bih[2][2], bhh[2][2];
    float biasv[2];
#pragma unroll
    for (int ti = 0; ti < 2; ++ti) {
        const int n0 = (wave + ti * 8) * 16;
#pragma unroll
        for (int kb = 0; kb < 2; ++kb) {
            bih[ti][kb] = load_bT_frag(W_ih, E_DIM, n0, kb * 32, lane);
            bhh[ti][kb] = load_bT_frag(W_hh, E_DIM, n0, kb * 32, lane);
        }
        const int col = n0 + (lane & 15);
        biasv[ti] = b_ih[col] + b_hh[col];
    }

    // ---- stage sequence indices, init state -------------------------------
    for (int i = tid; i < M_MEM * L_SEQ; i += 256)
        seq_lds[i] = act_seqs[gb * M_MEM * L_SEQ + i];
#pragma unroll
    for (int j = 0; j < 4; ++j) h_lds[m * XS + e4 + j] = (__bf16)0.0f;
    float c0 = 0.f, c1 = 0.f, c2 = 0.f, c3 = 0.f;
    __syncthreads();

    // prologue: kick off async gather of x_0
    {
        const int idx = seq_lds[m * L_SEQ + 0];
        async_copy_b128(&x_lds[0][m * XSF + e4],
                        act_table + (size_t)idx * E_DIM + e4);
    }

    // ======================= 50-step recurrent loop ========================
    for (int t = 0; t < L_SEQ; ++t) {
        // issue async gather of x_{t+1} into the other buffer (overlaps WMMA)
        if (t + 1 < L_SEQ) {
            const int idx2 = seq_lds[m * L_SEQ + t + 1];
            async_copy_b128(&x_lds[(t + 1) & 1][m * XSF + e4],
                            act_table + (size_t)idx2 * E_DIM + e4);
            wait_async_le1();            // x_t done; x_{t+1} may be in flight
        } else {
            wait_async_0();
        }
        __syncthreads();

        // gates[16,256] = bias + x @ W_ih^T + h @ W_hh^T   (WMMA bf16->f32)
        {
            const float* xb = x_lds[t & 1];
            const v16bf ax0 = load_a_frag_f32 (xb,    0,  lane);
            const v16bf ax1 = load_a_frag_f32 (xb,    32, lane);
            const v16bf ah0 = load_a_frag_bf16(h_lds, 0,  lane);
            const v16bf ah1 = load_a_frag_bf16(h_lds, 32, lane);
#pragma unroll
            for (int ti = 0; ti < 2; ++ti) {
                v8f acc;
#pragma unroll
                for (int r = 0; r < 8; ++r) acc[r] = biasv[ti];
                acc = wmma_bf16(ax0, bih[ti][0], acc);
                acc = wmma_bf16(ax1, bih[ti][1], acc);
                acc = wmma_bf16(ah0, bhh[ti][0], acc);
                acc = wmma_bf16(ah1, bhh[ti][1], acc);
                const int n     = (wave + ti * 8) * 16 + (lane & 15);
                const int rbase = (lane & 16) ? 8 : 0;
#pragma unroll
                for (int r = 0; r < 8; ++r)
                    gates[(rbase + r) * GSTR + n] = acc[r];
            }
        }
        __syncthreads();

        // cell update: c = sig(f)*c + sig(i)*tanh(g); h = sig(o)*tanh(c)
        {
            const float4 gi = *(const float4*)&gates[m * GSTR +   0 + e4];
            const float4 gf = *(const float4*)&gates[m * GSTR +  64 + e4];
            const float4 gg = *(const float4*)&gates[m * GSTR + 128 + e4];
            const float4 go = *(const float4*)&gates[m * GSTR + 192 + e4];
            c0 = sigmoidf_(gf.x) * c0 + sigmoidf_(gi.x) * tanhf_(gg.x);
            c1 = sigmoidf_(gf.y) * c1 + sigmoidf_(gi.y) * tanhf_(gg.y);
            c2 = sigmoidf_(gf.z) * c2 + sigmoidf_(gi.z) * tanhf_(gg.z);
            c3 = sigmoidf_(gf.w) * c3 + sigmoidf_(gi.w) * tanhf_(gg.w);
            const float h0 = sigmoidf_(go.x) * tanhf_(c0);
            const float h1 = sigmoidf_(go.y) * tanhf_(c1);
            const float h2 = sigmoidf_(go.z) * tanhf_(c2);
            const float h3 = sigmoidf_(go.w) * tanhf_(c3);
            h_lds[m * XS + e4 + 0] = (__bf16)h0;
            h_lds[m * XS + e4 + 1] = (__bf16)h1;
            h_lds[m * XS + e4 + 2] = (__bf16)h2;
            h_lds[m * XS + e4 + 3] = (__bf16)h3;
            if (t == L_SEQ - 1) {          // m_act -> fus accumulator
                fus[m * E_DIM + e4 + 0] = h0;
                fus[m * E_DIM + e4 + 1] = h1;
                fus[m * E_DIM + e4 + 2] = h2;
                fus[m * E_DIM + e4 + 3] = h3;
            }
        }
        __syncthreads();
    }

    // ======================= GCN: mean-aggregate + GEMM ====================
    {
        float a0 = 0.f, a1 = 0.f, a2 = 0.f, a3 = 0.f;
        const int base = gb * M_MEM + m;
        int idx = members[base];
#pragma unroll 1
        for (int j = 0; j < K_NBR + 1; ++j) {
            const float4 v = *(const float4*)(user_table + (size_t)idx * E_DIM + e4);
            if (j < K_NBR) {
                idx = neighbors[base * K_NBR + j];
                __builtin_prefetch(user_table + (size_t)idx * E_DIM + e4, 0, 1);
            }
            a0 += v.x; a1 += v.y; a2 += v.z; a3 += v.w;
        }
        const float inv = 1.0f / (float)(K_NBR + 1);
        h_lds[m * XS + e4 + 0] = (__bf16)(a0 * inv);   // reuse h tile for agg
        h_lds[m * XS + e4 + 1] = (__bf16)(a1 * inv);
        h_lds[m * XS + e4 + 2] = (__bf16)(a2 * inv);
        h_lds[m * XS + e4 + 3] = (__bf16)(a3 * inv);
    }
    __syncthreads();

    if (wave < 4) {   // agg @ gcn_W^T : 4 column tiles, one per wave
        const v16bf a0f = load_a_frag_bf16(h_lds, 0,  lane);
        const v16bf a1f = load_a_frag_bf16(h_lds, 32, lane);
        const v16bf w0  = load_bT_frag(gcn_W, E_DIM, wave * 16, 0,  lane);
        const v16bf w1  = load_bT_frag(gcn_W, E_DIM, wave * 16, 32, lane);
        v8f acc;
#pragma unroll
        for (int r = 0; r < 8; ++r) acc[r] = 0.f;
        acc = wmma_bf16(a0f, w0, acc);
        acc = wmma_bf16(a1f, w1, acc);
        const int n     = wave * 16 + (lane & 15);
        const int rbase = (lane & 16) ? 8 : 0;
#pragma unroll
        for (int r = 0; r < 8; ++r) gates[(rbase + r) * GSTR + n] = acc[r];
    }
    __syncthreads();

    // fus = m_act + relu(m_gcn + gcn_b)
#pragma unroll
    for (int j = 0; j < 4; ++j) {
        const float g = gates[m * GSTR + e4 + j] + gcn_b[e4 + j];
        fus[m * E_DIM + e4 + j] += fmaxf(g, 0.f);
    }
    __syncthreads();

    // ======================= PNA set aggregation ===========================
    if (tid < E_DIM) {
        float s = 0.f, mx = -3.0e38f;
#pragma unroll
        for (int mm = 0; mm < M_MEM; ++mm) {
            const float v = fus[mm * E_DIM + tid];
            s += v; mx = fmaxf(mx, v);
        }
        const float mean = s * (1.0f / M_MEM);
        const float amp  = 1.0f;   // log(M+1)/log(AVG_M+1) with M == AVG_M
        feats[tid]       = mean;
        feats[ 64 + tid] = mx;
        feats[128 + tid] = mean * amp;
        feats[192 + tid] = mx * amp;
    }
    __syncthreads();

    if (tid < E_DIM) {
        float acc = pna_b1[tid];
        for (int k = 0; k < 4 * E_DIM; ++k) acc += feats[k] * pna_W1[k * E_DIM + tid];
        v1[tid] = fmaxf(acc, 0.f);
    }
    __syncthreads();
    if (tid < E_DIM) {
        float acc = pna_b2[tid];
        for (int k = 0; k < E_DIM; ++k) acc += v1[k] * pna_W2[k * E_DIM + tid];
        v2[tid] = fmaxf(acc, 0.f);
    }
    __syncthreads();
    if (tid < E_DIM) {
        float acc = pna_b3[tid];
        for (int k = 0; k < E_DIM; ++k) acc += v2[k] * pna_W3[k * E_DIM + tid];
        grp[tid]  = acc;
        acte[tid] = act_table[(size_t)act_inputs[gb] * E_DIM + tid];
    }
    __syncthreads();

    // ======================= predict head ==================================
    if (tid < 8) {
        float acc = pred_b1[tid];
        for (int k = 0; k < E_DIM; ++k) {
            acc += grp[k] * acte[k] * pred_W1[(      k) * 8 + tid];
            acc += grp[k]           * pred_W1[( 64 + k) * 8 + tid];
            acc += acte[k]          * pred_W1[(128 + k) * 8 + tid];
        }
        hred[tid] = fmaxf(acc, 0.f);
    }
    __syncthreads();
    if (tid == 0) {
        float acc = pred_b2[0];
#pragma unroll
        for (int j = 0; j < 8; ++j) acc += hred[j] * pred_W2[j];
        out[gb] = sigmoidf_(acc);
    }
}

extern "C" void kernel_launch(void* const* d_in, const int* in_sizes, int n_in,
                              void* d_out, int out_size, void* d_ws, size_t ws_size,
                              hipStream_t stream) {
    (void)in_sizes; (void)n_in; (void)d_ws; (void)ws_size; (void)out_size;
    const int*   members    = (const int*)  d_in[0];
    const int*   neighbors  = (const int*)  d_in[1];
    const int*   act_inputs = (const int*)  d_in[2];
    const int*   act_seqs   = (const int*)  d_in[3];
    const float* act_table  = (const float*)d_in[4];
    const float* user_table = (const float*)d_in[5];
    const float* W_ih       = (const float*)d_in[6];
    const float* W_hh       = (const float*)d_in[7];
    const float* b_ih       = (const float*)d_in[8];
    const float* b_hh       = (const float*)d_in[9];
    const float* gcn_W      = (const float*)d_in[10];
    const float* gcn_b      = (const float*)d_in[11];
    const float* pna_W1     = (const float*)d_in[12];
    const float* pna_b1     = (const float*)d_in[13];
    const float* pna_W2     = (const float*)d_in[14];
    const float* pna_b2     = (const float*)d_in[15];
    const float* pna_W3     = (const float*)d_in[16];
    const float* pna_b3     = (const float*)d_in[17];
    const float* pred_W1    = (const float*)d_in[18];
    const float* pred_b1    = (const float*)d_in[19];
    const float* pred_W2    = (const float*)d_in[20];
    const float* pred_b2    = (const float*)d_in[21];
    float* out = (float*)d_out;

    fused_lstm_gcn_pna_pred<<<dim3(B_GROUPS), dim3(256), 0, stream>>>(
        members, neighbors, act_inputs, act_seqs, act_table, user_table,
        W_ih, W_hh, b_ih, b_hh, gcn_W, gcn_b,
        pna_W1, pna_b1, pna_W2, pna_b2, pna_W3, pna_b3,
        pred_W1, pred_b1, pred_W2, pred_b2, out);
}
